// LoRALinear_17188459119051
// MI455X (gfx1250) — matrix operand order repaired and checked by reading, not compiled, not run on hardware
//
#include <hip/hip_runtime.h>

typedef __attribute__((ext_vector_type(16))) __bf16 v16bf;
typedef __attribute__((ext_vector_type(8)))  __bf16 v8bf;
typedef __attribute__((ext_vector_type(8)))  float  v8f;
typedef __attribute__((ext_vector_type(4)))  unsigned int v4u;
typedef __attribute__((ext_vector_type(8)))  int v8i;
typedef __attribute__((ext_vector_type(4)))  int v4i;

#define T_DIM 8192
#define N_DIM 4096
#define M_DIM 4096
#define RANK  8
#define ALPHA 1.0f          // 8.0 / RANK
#define BK    32
#define NK    (N_DIM / BK)
#define LDS_STRIDE 40       // 32 + 8 halves pad: 16B-aligned chunks, spreads LDS banks

// ----------------------------- bf16 split helpers -----------------------------
__device__ __forceinline__ unsigned short f32_to_bf16_rn(float f) {
    unsigned u = __float_as_uint(f);
    u += 0x7FFFu + ((u >> 16) & 1u);          // round to nearest even
    return (unsigned short)(u >> 16);
}
__device__ __forceinline__ float bf16_us_to_f32(unsigned short h) {
    return __uint_as_float(((unsigned)h) << 16);
}
__device__ __forceinline__ void split_bf16(float f, unsigned short& hi, unsigned short& lo) {
    hi = f32_to_bf16_rn(f);
    lo = f32_to_bf16_rn(f - bf16_us_to_f32(hi));
}
__device__ __forceinline__ unsigned pack2(unsigned short a, unsigned short b) {
    return (unsigned)a | ((unsigned)b << 16);
}
__device__ __forceinline__ void split4(float4 f, uint2& hi, uint2& lo) {
    unsigned short h0, h1, h2, h3, l0, l1, l2, l3;
    split_bf16(f.x, h0, l0); split_bf16(f.y, h1, l1);
    split_bf16(f.z, h2, l2); split_bf16(f.w, h3, l3);
    hi = make_uint2(pack2(h0, h1), pack2(h2, h3));
    lo = make_uint2(pack2(l0, l1), pack2(l2, l3));
}

union FragA { v16bf v; v8bf h[2]; };

// --------------------- shared WMMA compute + epilogue ---------------------
__device__ __forceinline__ void compute_step(
    const unsigned short* sxh_, const unsigned short* sxl_,
    const unsigned short* swh_, const unsigned short* swl_,
    int wt, int wm, int col, int kA, int kB, v8f (&acc)[4][2])
{
    FragA bh[2], bl[2];
#pragma unroll
    for (int tj = 0; tj < 2; ++tj) {
        const unsigned short* pb = &swh_[(wm * 32 + tj * 16 + col) * LDS_STRIDE + kB];
        const unsigned short* ql = &swl_[(wm * 32 + tj * 16 + col) * LDS_STRIDE + kB];
        bh[tj].h[0] = *reinterpret_cast<const v8bf*>(pb);
        bh[tj].h[1] = *reinterpret_cast<const v8bf*>(pb + 16);
        bl[tj].h[0] = *reinterpret_cast<const v8bf*>(ql);
        bl[tj].h[1] = *reinterpret_cast<const v8bf*>(ql + 16);
    }
#pragma unroll
    for (int ti = 0; ti < 4; ++ti) {
        const unsigned short* pa = &sxh_[(wt * 64 + ti * 16 + col) * LDS_STRIDE];
        const unsigned short* qa = &sxl_[(wt * 64 + ti * 16 + col) * LDS_STRIDE];
        FragA ah, al;
        ah.h[0] = *reinterpret_cast<const v8bf*>(pa + kA);
        ah.h[1] = *reinterpret_cast<const v8bf*>(pa + kA + 16);
        al.h[0] = *reinterpret_cast<const v8bf*>(qa + kA);
        al.h[1] = *reinterpret_cast<const v8bf*>(qa + kA + 16);
#pragma unroll
        for (int tj = 0; tj < 2; ++tj) {
            acc[ti][tj] = __builtin_amdgcn_wmma_f32_16x16x32_bf16(
                false, ah.v, false, bh[tj].v, (short)0, acc[ti][tj], false, false);
            acc[ti][tj] = __builtin_amdgcn_wmma_f32_16x16x32_bf16(
                false, ah.v, false, bl[tj].v, (short)0, acc[ti][tj], false, false);
            acc[ti][tj] = __builtin_amdgcn_wmma_f32_16x16x32_bf16(
                false, al.v, false, bh[tj].v, (short)0, acc[ti][tj], false, false);
        }
    }
}

__device__ __forceinline__ void epilogue_store(
    const float (*As_)[RANK], const float (*xBs_)[RANK], const float* bs_,
    int wt, int wm, int col, int lane, int tBase, int mBase,
    v8f (&acc)[4][2], float* __restrict__ out)
{
    float Ar[2][RANK];
    float bb[2];
#pragma unroll
    for (int tj = 0; tj < 2; ++tj) {
        const int ml = wm * 32 + tj * 16 + col;
        bb[tj] = bs_[ml];
#pragma unroll
        for (int r = 0; r < RANK; ++r) Ar[tj][r] = As_[ml][r];
    }
#pragma unroll
    for (int ti = 0; ti < 4; ++ti) {
#pragma unroll
        for (int i = 0; i < 8; ++i) {
            const int tl = wt * 64 + ti * 16 + (lane >> 4) * 8 + i;  // C/D: M = i + (L/16)*8
            float xr[RANK];
#pragma unroll
            for (int r = 0; r < RANK; ++r) xr[r] = xBs_[tl][r];
#pragma unroll
            for (int tj = 0; tj < 2; ++tj) {
                float v = acc[ti][tj][i] + bb[tj];
#pragma unroll
                for (int r = 0; r < RANK; ++r) v += xr[r] * Ar[tj][r];
                out[(size_t)(tBase + tl) * M_DIM + (mBase + wm * 32 + tj * 16 + col)] = v;
            }
        }
    }
}

// ---------------- kernel 1: xB[t][r] = sum_n x[t,n] * B[n,r] ----------------
__global__ __launch_bounds__(256) void xb_kernel(
    const float* __restrict__ x, const float* __restrict__ B, float* __restrict__ xB)
{
    __shared__ float red[256][RANK];
    const int t = blockIdx.x;
    const int tid = threadIdx.x;
    const int n0 = tid * 16;                       // 256 threads * 16 = 4096 = N
    const float* xrow = x + (size_t)t * N_DIM;
    float acc[RANK] = {};
#pragma unroll
    for (int j = 0; j < 16; ++j) {
        const float xv = xrow[n0 + j];
        const float4 b0 = *reinterpret_cast<const float4*>(&B[(size_t)(n0 + j) * RANK]);
        const float4 b1 = *reinterpret_cast<const float4*>(&B[(size_t)(n0 + j) * RANK + 4]);
        acc[0] += xv * b0.x; acc[1] += xv * b0.y; acc[2] += xv * b0.z; acc[3] += xv * b0.w;
        acc[4] += xv * b1.x; acc[5] += xv * b1.y; acc[6] += xv * b1.z; acc[7] += xv * b1.w;
    }
#pragma unroll
    for (int r = 0; r < RANK; ++r) red[tid][r] = acc[r];
    __syncthreads();
    for (int s = 128; s > 0; s >>= 1) {
        if (tid < s) {
#pragma unroll
            for (int r = 0; r < RANK; ++r) red[tid][r] += red[tid + s][r];
        }
        __syncthreads();
    }
    if (tid < RANK) xB[(size_t)t * RANK + tid] = red[0][tid];
}

// -------- kernel 2: one-time fp32 -> (bf16 hi, bf16 lo) split pre-pass --------
__global__ __launch_bounds__(256) void split_kernel(
    const float* __restrict__ src, unsigned short* __restrict__ dh,
    unsigned short* __restrict__ dl, size_t n4)
{
    size_t i = (size_t)blockIdx.x * 256 + threadIdx.x;
    const size_t stride = (size_t)gridDim.x * 256;
    for (; i < n4; i += stride) {
        const float4 v = reinterpret_cast<const float4*>(src)[i];
        uint2 hi, lo;
        split4(v, hi, lo);
        reinterpret_cast<uint2*>(dh)[i] = hi;
        reinterpret_cast<uint2*>(dl)[i] = lo;
    }
}

// --------------------------- TDM descriptor issue ---------------------------
// 2-D tile: 128 rows x 32 bf16, row stride 4096 bf16 in memory; LDS padding of
// 4 DWORDs after every 16 DWORDs reproduces the 40-half LDS row stride.
__device__ __forceinline__ void tdm_load_tile(const unsigned short* gsrc, unsigned ldsAddr)
{
    const unsigned long long ga = (unsigned long long)gsrc;
    const v4u g0 = { 1u,                                   // count=1, user mode
                     ldsAddr,                              // LDS byte address
                     (unsigned)ga,                         // global_addr[31:0]
                     (unsigned)((ga >> 32) & 0x01FFFFFFu) | (2u << 30) }; // addr[56:32] | type=2
    const v8i g1 = {
        (int)((1u << 16) | (1u << 20) | (3u << 22) | (3u << 25)), // data_size=2B, pad_en, pad_interval=16DW, pad_amount=4DW
        (int)(0x1000u << 16),   // tensor_dim0 = 4096 (bits 79:48, low half here)
        (int)(0x2000u << 16),   // tensor_dim0 hi=0 | tensor_dim1 = 8192 (low 16)
        (int)(32u << 16),       // tensor_dim1 hi=0 | tile_dim0 = 32
        128,                    // tile_dim1 = 128 | tile_dim2 = 0
        4096,                   // tensor_dim0_stride (low 32) = 4096 elements
        0, 0 };
    const v4i z4 = { 0, 0, 0, 0 };
    const v8i z8 = { 0, 0, 0, 0, 0, 0, 0, 0 };
    __builtin_amdgcn_tensor_load_to_lds(g0, g1, z4, z4, z8, 0);
}

// ---------------- kernel 3a: GEMM, TDM double-buffered bf16 tiles ----------------
__global__ __launch_bounds__(256, 1) void lora_gemm_tdm_kernel(
    const unsigned short* __restrict__ xh, const unsigned short* __restrict__ xl,
    const unsigned short* __restrict__ wh, const unsigned short* __restrict__ wl,
    const float* __restrict__ bvec, const float* __restrict__ A,
    const float* __restrict__ xB, float* __restrict__ out)
{
    __shared__ __align__(16) unsigned short sxh[2][128 * LDS_STRIDE];
    __shared__ __align__(16) unsigned short sxl[2][128 * LDS_STRIDE];
    __shared__ __align__(16) unsigned short swh[2][128 * LDS_STRIDE];
    __shared__ __align__(16) unsigned short swl[2][128 * LDS_STRIDE];
    __shared__ __align__(16) float As[128][RANK];
    __shared__ __align__(16) float xBs[128][RANK];
    __shared__ float bs[128];

    const int tid  = threadIdx.x;
    const int lane = tid & 31;
    const int wave = tid >> 5;
    const int wt = wave >> 2;
    const int wm = wave & 3;
    const int tBase = blockIdx.y * 128;
    const int mBase = blockIdx.x * 128;

    // stage epilogue operands
    {
        const int r = tid >> 1, p = (tid & 1) * 4;
        float4 a4 = *reinterpret_cast<const float4*>(&A[(size_t)(mBase + r) * RANK + p]);
        a4.x *= ALPHA; a4.y *= ALPHA; a4.z *= ALPHA; a4.w *= ALPHA;
        *reinterpret_cast<float4*>(&As[r][p]) = a4;
        *reinterpret_cast<float4*>(&xBs[r][p]) =
            *reinterpret_cast<const float4*>(&xB[(size_t)(tBase + r) * RANK + p]);
        if (tid < 128) bs[tid] = bvec[mBase + tid];
    }

    const unsigned short* gxh = xh + (size_t)tBase * N_DIM;
    const unsigned short* gxl = xl + (size_t)tBase * N_DIM;
    const unsigned short* gwh = wh + (size_t)mBase * N_DIM;
    const unsigned short* gwl = wl + (size_t)mBase * N_DIM;

    // prologue: bring K-tile 0 into buffer 0 via the Tensor Data Mover
    if (wave == 0) {
        tdm_load_tile(gxh, (unsigned)(unsigned long long)&sxh[0][0]);
        tdm_load_tile(gxl, (unsigned)(unsigned long long)&sxl[0][0]);
        tdm_load_tile(gwh, (unsigned)(unsigned long long)&swh[0][0]);
        tdm_load_tile(gwl, (unsigned)(unsigned long long)&swl[0][0]);
        __builtin_amdgcn_s_wait_tensorcnt(0);
    }
    __syncthreads();

    v8f acc[4][2] = {};
    const int col = lane & 15;
    const int kA  = (lane >> 4) * 8;
    const int kB  = (lane >> 4) * 8;

    for (int kt = 0; kt < NK; ++kt) {
        const int cur = kt & 1;
        const int nxt = cur ^ 1;
        const bool more = (kt + 1 < NK);
        if (wave == 0 && more) {        // async: TDM fills the other buffer
            const size_t off = (size_t)(kt + 1) * BK;
            tdm_load_tile(gxh + off, (unsigned)(unsigned long long)&sxh[nxt][0]);
            tdm_load_tile(gxl + off, (unsigned)(unsigned long long)&sxl[nxt][0]);
            tdm_load_tile(gwh + off, (unsigned)(unsigned long long)&swh[nxt][0]);
            tdm_load_tile(gwl + off, (unsigned)(unsigned long long)&swl[nxt][0]);
        }
        compute_step(sxh[cur], sxl[cur], swh[cur], swl[cur], wt, wm, col, kA, kB, acc);
        if (wave == 0 && more) __builtin_amdgcn_s_wait_tensorcnt(0);
        __syncthreads();
    }

    epilogue_store(As, xBs, bs, wt, wm, col, lane, tBase, mBase, acc, out);
}

// ------- kernel 3b: fallback GEMM (in-kernel split), used if d_ws is small -------
__global__ __launch_bounds__(256, 1) void lora_gemm_fused_kernel(
    const float* __restrict__ x, const float* __restrict__ W,
    const float* __restrict__ bvec, const float* __restrict__ A,
    const float* __restrict__ xB, float* __restrict__ out)
{
    __shared__ __align__(16) unsigned short sxh[128 * LDS_STRIDE];
    __shared__ __align__(16) unsigned short sxl[128 * LDS_STRIDE];
    __shared__ __align__(16) unsigned short swh[128 * LDS_STRIDE];
    __shared__ __align__(16) unsigned short swl[128 * LDS_STRIDE];
    __shared__ __align__(16) float As[128][RANK];
    __shared__ __align__(16) float xBs[128][RANK];
    __shared__ float bs[128];

    const int tid  = threadIdx.x;
    const int lane = tid & 31;
    const int wave = tid >> 5;
    const int wt = wave >> 2;
    const int wm = wave & 3;
    const int tBase = blockIdx.y * 128;
    const int mBase = blockIdx.x * 128;

    {
        const int r = tid >> 1, p = (tid & 1) * 4;
        float4 a4 = *reinterpret_cast<const float4*>(&A[(size_t)(mBase + r) * RANK + p]);
        a4.x *= ALPHA; a4.y *= ALPHA; a4.z *= ALPHA; a4.w *= ALPHA;
        *reinterpret_cast<float4*>(&As[r][p]) = a4;
        *reinterpret_cast<float4*>(&xBs[r][p]) =
            *reinterpret_cast<const float4*>(&xB[(size_t)(tBase + r) * RANK + p]);
        if (tid < 128) bs[tid] = bvec[mBase + tid];
    }

    v8f acc[4][2] = {};
    float4 rx[4], rw[4];
#pragma unroll
    for (int j = 0; j < 4; ++j) {
        const int f = tid + j * 256;
        const int row = f >> 3, kq = (f & 7) * 4;
        rx[j] = *reinterpret_cast<const float4*>(&x[(size_t)(tBase + row) * N_DIM + kq]);
        rw[j] = *reinterpret_cast<const float4*>(&W[(size_t)(mBase + row) * N_DIM + kq]);
    }

    const int col = lane & 15;
    const int kA  = (lane >> 4) * 8;
    const int kB  = (lane >> 4) * 8;

    for (int kt = 0; kt < NK; ++kt) {
        __syncthreads();
#pragma unroll
        for (int j = 0; j < 4; ++j) {
            const int f = tid + j * 256;
            const int row = f >> 3, kq = (f & 7) * 4;
            uint2 hi, lo;
            split4(rx[j], hi, lo);
            *reinterpret_cast<uint2*>(&sxh[row * LDS_STRIDE + kq]) = hi;
            *reinterpret_cast<uint2*>(&sxl[row * LDS_STRIDE + kq]) = lo;
            split4(rw[j], hi, lo);
            *reinterpret_cast<uint2*>(&swh[row * LDS_STRIDE + kq]) = hi;
            *reinterpret_cast<uint2*>(&swl[row * LDS_STRIDE + kq]) = lo;
        }
        if (kt + 1 < NK) {
            const int kNext = (kt + 1) * BK;
#pragma unroll
            for (int j = 0; j < 4; ++j) {
                const int f = tid + j * 256;
                const int row = f >> 3, kq = (f & 7) * 4;
                rx[j] = *reinterpret_cast<const float4*>(&x[(size_t)(tBase + row) * N_DIM + kNext + kq]);
                rw[j] = *reinterpret_cast<const float4*>(&W[(size_t)(mBase + row) * N_DIM + kNext + kq]);
            }
        }
        __syncthreads();
        compute_step(sxh, sxl, swh, swl, wt, wm, col, kA, kB, acc);
    }

    epilogue_store(As, xBs, bs, wt, wm, col, lane, tBase, mBase, acc, out);
}

extern "C" void kernel_launch(void* const* d_in, const int* in_sizes, int n_in,
                              void* d_out, int out_size, void* d_ws, size_t ws_size,
                              hipStream_t stream) {
    (void)in_sizes; (void)n_in; (void)out_size;
    const float* x = (const float*)d_in[0];
    const float* W = (const float*)d_in[1];
    const float* b = (const float*)d_in[2];
    const float* A = (const float*)d_in[3];
    const float* B = (const float*)d_in[4];
    float* out = (float*)d_out;

    // workspace layout: [xB 256KB][x_hi 64MB][x_lo 64MB][w_hi 32MB][w_lo 32MB]
    float* xB = (float*)d_ws;
    unsigned short* xh = (unsigned short*)((char*)d_ws + (1 << 18));
    unsigned short* xl = xh + (size_t)T_DIM * N_DIM;
    unsigned short* wh = xl + (size_t)T_DIM * N_DIM;
    unsigned short* wl = wh + (size_t)M_DIM * N_DIM;
    const size_t need = (size_t)(1 << 18)
                      + 2 * (size_t)T_DIM * N_DIM * sizeof(unsigned short)
                      + 2 * (size_t)M_DIM * N_DIM * sizeof(unsigned short);

    xb_kernel<<<T_DIM, 256, 0, stream>>>(x, B, xB);
    dim3 grid(M_DIM / 128, T_DIM / 128);

    if (ws_size >= need) {
        split_kernel<<<4096, 256, 0, stream>>>(x, xh, xl, (size_t)T_DIM * N_DIM / 4);
        split_kernel<<<4096, 256, 0, stream>>>(W, wh, wl, (size_t)M_DIM * N_DIM / 4);
        lora_gemm_tdm_kernel<<<grid, 256, 0, stream>>>(xh, xl, wh, wl, b, A, xB, out);
    } else {
        lora_gemm_fused_kernel<<<grid, 256, 0, stream>>>(x, W, b, A, xB, out);
    }
}
